// WeatherModel_12103217840138
// MI455X (gfx1250) — compile-verified, matrix operand order checked by reading, not run on hardware
//
#include <hip/hip_runtime.h>
#include <hip/hip_bf16.h>

// ---------------------------------------------------------------------------
// ConvLSTM encoder-decoder for MI455X (gfx1250).
// Big convs (Cin 65/128 -> Cout 256) run as implicit GEMM through
// v_wmma_f32_16x16x32_bf16:
//  - channel-concat input materialized contiguously with guard pads so the
//    im2col gather is a single-base, clamp-free load
//  - tap validity applied by bit-masking the loaded float (no select, no
//    EXEC predication); k >= KT region needs no mask because the packed B
//    fragments are zero there
//  - 2-deep software pipeline, manually unrolled x2 (ping/pong operand sets)
// ---------------------------------------------------------------------------

typedef __attribute__((ext_vector_type(16))) __bf16 v16bf;
typedef __attribute__((ext_vector_type(8)))  float  v8f;

#define SP   9216      // 96*96
#define BB   4
#define TT   10
#define DD   5
#define HH   64
#define MG   (BB*SP)   // 36864 GEMM rows
#define COUT 256

union AFrag { v16bf v; __bf16 h[16]; unsigned short u[16]; };
struct Bset { v16bf b0, b1, b2, b3; };

__device__ __forceinline__ unsigned short f2bf(float f) {
  unsigned u = __float_as_uint(f);
  unsigned r = u + 0x7FFFu + ((u >> 16) & 1u);   // round-to-nearest-even
  return (unsigned short)(r >> 16);
}

__device__ __forceinline__ float sigm(float x) { return 1.f / (1.f + __expf(-x)); }

// ---------------------------------------------------------------------------
// Pack OIHW f32 weights into bf16 B-fragments:
// wB[chunk][tile(16)][lane(32)][half(16)], tile covers 16 output channels.
// k >= KT rows are zero-filled (this also licenses the unmasked A gather).
// ---------------------------------------------------------------------------
__global__ void pack_wB(const float* __restrict__ w, int CIN, int KT,
                        unsigned short* __restrict__ wB, int total) {
  int i = blockIdx.x * blockDim.x + threadIdx.x;
  if (i >= total) return;
  int half = i & 15;
  int lane = (i >> 4) & 31;
  int tile = (i >> 9) & 15;
  int chunk = i >> 13;
  int n = tile * 16 + (lane & 15);
  int kbase = (lane < 16) ? 0 : 8;
  int kin = kbase + (half & 7) + ((half >> 3) << 4);
  int k = chunk * 32 + kin;
  unsigned short hv = 0;
  if (k < KT) {
    int ci = k / 9;
    int tap = k - ci * 9;
    int ky = tap / 3;
    int kx = tap - ky * 3;
    hv = f2bf(w[((n * CIN + ci) * 3 + ky) * 3 + kx]);
  }
  wB[i] = hv;
}

// ---------------------------------------------------------------------------
// Materialize concat(xA, xB) -> contiguous (b, CIN, SP) buffer for the GEMM.
// ---------------------------------------------------------------------------
__global__ void concat_copy(const float* __restrict__ xA, int cinA,
                            const float* __restrict__ xB, int cinB,
                            float* __restrict__ dst, int total) {
  int i = blockIdx.x * blockDim.x + threadIdx.x;
  if (i >= total) return;
  int CIN = cinA + cinB;
  int b = i / (CIN * SP);
  int r = i - b * CIN * SP;
  int ci = r / SP;
  int s = r - ci * SP;
  dst[i] = (ci < cinA) ? xA[(b * cinA + ci) * SP + s]
                       : xB[(b * cinB + (ci - cinA)) * SP + s];
}

// ---------------------------------------------------------------------------
// Branch-free im2col A-fragment gather against a guard-padded contiguous
// input. Tap validity is applied by AND-masking the loaded float's bits,
// so the load has an unconditional use and cannot be EXEC-predicated.
// ---------------------------------------------------------------------------
__device__ __forceinline__ AFrag gatherA(int c, int kbase,
                                         const float* __restrict__ xg,
                                         int idxBase, unsigned mask9,
                                         int CINm1) {
  AFrag a;
#pragma unroll
  for (int j = 0; j < 16; ++j) {
    int k = c * 32 + kbase + (j & 7) + ((j >> 3) << 4);
    int ci = k / 9;
    int tap = k - ci * 9;
    int ky = tap / 3;
    int kx = tap - ky * 3;
    int cic = min(ci, CINm1);               // address safety for k >= KT tail
    float v = xg[idxBase + cic * SP + ky * 96 + kx];  // guard pads make safe
    unsigned msk = (unsigned)(-(int)((mask9 >> tap) & 1u));  // all-ones / 0
    float vm = __uint_as_float(__float_as_uint(v) & msk);
    a.h[j] = (__bf16)vm;
  }
  return a;
}

__device__ __forceinline__ Bset loadB(const unsigned short* __restrict__ wbase,
                                      int c) {
  const unsigned short* wp = wbase + (size_t)c * 8192;
  Bset s;
  s.b0 = *(const v16bf*)(wp + 0 * 512);
  s.b1 = *(const v16bf*)(wp + 1 * 512);
  s.b2 = *(const v16bf*)(wp + 2 * 512);
  s.b3 = *(const v16bf*)(wp + 3 * 512);
  return s;
}

// ---------------------------------------------------------------------------
// Implicit-GEMM ConvLSTM pre-activation: z[m][n] = conv3x3(xin)[n]
// Block: 256 threads = 8 waves; tile 128(M) x 64(N); grid (288, 4).
// Each wave owns 16 rows, 4 WMMA accumulators along N. 2-deep pipeline,
// manual x2 unroll with ping/pong operand sets.
// ---------------------------------------------------------------------------
__global__ __launch_bounds__(256)
void convlstm_gemm_bf16(const float* __restrict__ xin, int CIN,
                        const unsigned short* __restrict__ wB,
                        int KC,
                        float* __restrict__ z) {
  const int lane = threadIdx.x & 31;
  const int wave = threadIdx.x >> 5;
  const int mrow = blockIdx.x * 128 + wave * 16 + (lane & 15);
  const int nblk = blockIdx.y;
  const int b = mrow / SP;
  const int s = mrow - b * SP;
  const int y = s / 96;
  const int x = s - y * 96;
  const int kbase = (lane < 16) ? 0 : 8;
  const int CINm1 = CIN - 1;
  const int idxBase = b * CIN * SP + (y - 1) * 96 + (x - 1);
  const unsigned short* wbase = wB + ((size_t)(nblk * 4) * 32 + lane) * 16;

  // 9-tap border validity, one bit per tap
  unsigned mask9 = 0;
#pragma unroll
  for (int tap = 0; tap < 9; ++tap) {
    int ky = tap / 3, kx = tap % 3;                  // compile-time
    unsigned bit = ((unsigned)(y + ky - 1) < 96u) &
                   ((unsigned)(x + kx - 1) < 96u);
    mask9 |= bit << tap;
  }

  v8f acc0 = {}, acc1 = {}, acc2 = {}, acc3 = {};

  // pipeline prologue: chunks 0 and 1 into ping/pong sets
  AFrag a0 = gatherA(0, kbase, xin, idxBase, mask9, CINm1);
  Bset  s0 = loadB(wbase, 0);
  int c1i = (KC > 1) ? 1 : 0;
  AFrag a1 = gatherA(c1i, kbase, xin, idxBase, mask9, CINm1);
  Bset  s1 = loadB(wbase, c1i);

  for (int c = 0; c < KC; c += 2) {
    // chunk c (set0)
    acc0 = __builtin_amdgcn_wmma_f32_16x16x32_bf16(false, a0.v, false, s0.b0,
                                                   (short)0, acc0, false, false);
    acc1 = __builtin_amdgcn_wmma_f32_16x16x32_bf16(false, a0.v, false, s0.b1,
                                                   (short)0, acc1, false, false);
    acc2 = __builtin_amdgcn_wmma_f32_16x16x32_bf16(false, a0.v, false, s0.b2,
                                                   (short)0, acc2, false, false);
    acc3 = __builtin_amdgcn_wmma_f32_16x16x32_bf16(false, a0.v, false, s0.b3,
                                                   (short)0, acc3, false, false);
    int cf = (c + 2 < KC) ? c + 2 : c;               // tail refetch harmless
    a0 = gatherA(cf, kbase, xin, idxBase, mask9, CINm1);
    s0 = loadB(wbase, cf);
    if (c + 1 < KC) {
      // chunk c+1 (set1)
      acc0 = __builtin_amdgcn_wmma_f32_16x16x32_bf16(false, a1.v, false, s1.b0,
                                                     (short)0, acc0, false, false);
      acc1 = __builtin_amdgcn_wmma_f32_16x16x32_bf16(false, a1.v, false, s1.b1,
                                                     (short)0, acc1, false, false);
      acc2 = __builtin_amdgcn_wmma_f32_16x16x32_bf16(false, a1.v, false, s1.b2,
                                                     (short)0, acc2, false, false);
      acc3 = __builtin_amdgcn_wmma_f32_16x16x32_bf16(false, a1.v, false, s1.b3,
                                                     (short)0, acc3, false, false);
      int cg = (c + 3 < KC) ? c + 3 : c + 1;
      a1 = gatherA(cg, kbase, xin, idxBase, mask9, CINm1);
      s1 = loadB(wbase, cg);
    }
  }

  // C/D layout: VGPR r -> M = r (lanes 0-15) or r+8 (lanes 16-31), N = lane%16
  int mbase = blockIdx.x * 128 + wave * 16 + ((lane < 16) ? 0 : 8);
  int ncol = nblk * 64 + (lane & 15);
#pragma unroll
  for (int r = 0; r < 8; ++r) {
    float* zp = z + (size_t)(mbase + r) * COUT + ncol;
    zp[0]  = acc0[r];
    zp[16] = acc1[r];
    zp[32] = acc2[r];
    zp[48] = acc3[r];
  }
}

// ---------------------------------------------------------------------------
// LSTM gate pointwise: c' = sig(f)*c + sig(i)*tanh(g); h' = sig(o)*tanh(c').
// zLayout: 1 -> z[m][4CH] (GEMM output), 0 -> z[b][4CH][SP] (direct conv).
// ---------------------------------------------------------------------------
__global__ void lstm_gates(const float* __restrict__ z,
                           const float* __restrict__ bias, int CH, int zLayout,
                           float* __restrict__ hOut, long hBS,
                           float* __restrict__ cState, int total) {
  int i = blockIdx.x * blockDim.x + threadIdx.x;
  if (i >= total) return;
  int b = i / (CH * SP);
  int r = i - b * CH * SP;
  int ch = r / SP;
  int s = r - ch * SP;
  float zi, zf, zo, zg;
  if (zLayout) {
    long m = (long)(b * SP + s) * COUT;
    zi = z[m + ch];
    zf = z[m + CH + ch];
    zo = z[m + 2 * CH + ch];
    zg = z[m + 3 * CH + ch];
  } else {
    long base = (long)b * 4 * CH * SP + s;
    zi = z[base + (long)ch * SP];
    zf = z[base + (long)(CH + ch) * SP];
    zo = z[base + (long)(2 * CH + ch) * SP];
    zg = z[base + (long)(3 * CH + ch) * SP];
  }
  zi += bias[ch];
  zf += bias[CH + ch];
  zo += bias[2 * CH + ch];
  zg += bias[3 * CH + ch];
  long cidx = (long)(b * CH + ch) * SP + s;
  float c = cState[cidx];
  c = sigm(zf) * c + sigm(zi) * tanhf(zg);
  float h = sigm(zo) * tanhf(c);
  cState[cidx] = c;
  hOut[(long)b * hBS + (long)ch * SP + s] = h;
}

// ---------------------------------------------------------------------------
// Generic direct 3x3 conv (small layers / attention); optional concat input,
// optional additive tensor before activation, act: 0=none, 1=tanh.
// ---------------------------------------------------------------------------
__global__ void conv3x3_direct(const float* __restrict__ xA, long bsA, long csA,
                               int cinA, const float* __restrict__ xB, long bsB,
                               long csB, int cinB, const float* __restrict__ w,
                               const float* __restrict__ bias, int cout,
                               const float* __restrict__ addT, int act,
                               float* __restrict__ out, long bsO, long csO,
                               int total) {
  int i = blockIdx.x * blockDim.x + threadIdx.x;
  if (i >= total) return;
  int b = i / (cout * SP);
  int r = i - b * cout * SP;
  int co = r / SP;
  int s = r - co * SP;
  int y = s / 96;
  int x = s - y * 96;
  int CIN = cinA + cinB;
  float acc = bias ? bias[co] : 0.f;
  for (int ci = 0; ci < CIN; ++ci) {
    const float* src = (ci < cinA) ? (xA + (long)b * bsA + (long)ci * csA)
                                   : (xB + (long)b * bsB + (long)(ci - cinA) * csB);
    const float* wp = w + ((long)(co * CIN + ci)) * 9;
#pragma unroll
    for (int ky = 0; ky < 3; ++ky) {
      int yy = y + ky - 1;
      if (yy < 0 || yy >= 96) continue;
#pragma unroll
      for (int kx = 0; kx < 3; ++kx) {
        int xx = x + kx - 1;
        if (xx < 0 || xx >= 96) continue;
        acc += wp[ky * 3 + kx] * src[yy * 96 + xx];
      }
    }
  }
  if (addT) acc += addT[(long)(b * cout + co) * SP + s];
  if (act == 1) acc = tanhf(acc);
  out[(long)b * bsO + (long)co * csO + s] = acc;
}

// ---------------------------------------------------------------------------
// Attention helpers
// ---------------------------------------------------------------------------
__global__ void fill_f32(float* p, float v, long n) {
  long i = (long)blockIdx.x * blockDim.x + threadIdx.x;
  long st = (long)gridDim.x * blockDim.x;
  for (; i < n; i += st) p[i] = v;
}

__global__ void fill_u32(unsigned* p, unsigned v, int n) {
  int i = blockIdx.x * blockDim.x + threadIdx.x;
  if (i < n) p[i] = v;
}

__global__ void exp_max_reduce(const float* __restrict__ alpha, int n,
                               unsigned* __restrict__ gmax) {
  __shared__ float sm[256];
  float v = 0.f;
  for (int i = blockIdx.x * 256 + threadIdx.x; i < n; i += gridDim.x * 256)
    v = fmaxf(v, __expf(alpha[i]));
  sm[threadIdx.x] = v;
  __syncthreads();
  for (int st = 128; st > 0; st >>= 1) {
    if ((int)threadIdx.x < st)
      sm[threadIdx.x] = fmaxf(sm[threadIdx.x], sm[threadIdx.x + st]);
    __syncthreads();
  }
  if (threadIdx.x == 0) atomicMax(gmax, __float_as_uint(sm[0]));  // ea > 0
}

__global__ void attn_apply(float* __restrict__ xs, const float* __restrict__ alpha,
                           const unsigned* __restrict__ gmax, int total) {
  int i = blockIdx.x * blockDim.x + threadIdx.x;
  if (i >= total) return;
  int b = i / (TT * DD * SP);
  int r = i - b * TT * DD * SP;
  int t = r / (DD * SP);
  int r2 = r - t * DD * SP;
  int d = r2 / SP;
  int s = r2 - d * SP;
  (void)t;
  float mx = __uint_as_float(*gmax);
  xs[i] *= __expf(alpha[(b * DD + d) * SP + s]) / mx;
}

// ---------------------------------------------------------------------------
// Decoder output shift-register (11 channels) + 1x1 tcnn conv + relu.
// ---------------------------------------------------------------------------
__global__ void ring_init(const float* __restrict__ x, float* __restrict__ ring,
                          int total) {  // total = B*10*SP
  int i = blockIdx.x * blockDim.x + threadIdx.x;
  if (i >= total) return;
  int b = i / (TT * SP);
  int r = i - b * TT * SP;
  int t = r / SP;
  int s = r - t * SP;
  ring[((long)b * 11 + t) * SP + s] = x[((long)(b * TT + t) * DD + 0) * SP + s];  // SEL=0
}

__global__ void tcnn_step(const float* __restrict__ ring,
                          const float* __restrict__ w,
                          const float* __restrict__ bias, int p,
                          float* __restrict__ hState, float* __restrict__ outT,
                          long bsO, int total) {  // total = B*SP
  int i = blockIdx.x * blockDim.x + threadIdx.x;
  if (i >= total) return;
  int b = i / SP;
  int s = i - b * SP;
  float acc = bias[0];
#pragma unroll
  for (int j = 0; j < 11; ++j) {
    int slot = p + j;
    if (slot >= 11) slot -= 11;
    acc += w[j] * ring[((long)b * 11 + slot) * SP + s];
  }
  float h = fmaxf(acc, 0.f);
  hState[i] = h;
  outT[(long)b * bsO + s] = h;
}

// ---------------------------------------------------------------------------
// Host orchestration
// ---------------------------------------------------------------------------
extern "C" void kernel_launch(void* const* d_in, const int* in_sizes, int n_in,
                              void* d_out, int out_size, void* d_ws, size_t ws_size,
                              hipStream_t stream) {
  (void)in_sizes; (void)n_in; (void)out_size; (void)ws_size;
  const float* x       = (const float*)d_in[0];
  const float* enc_w0  = (const float*)d_in[1];
  const float* enc_b0  = (const float*)d_in[2];
  const float* enc_w1  = (const float*)d_in[3];
  const float* enc_b1  = (const float*)d_in[4];
  const float* enc_w2  = (const float*)d_in[5];
  const float* enc_b2  = (const float*)d_in[6];
  const float* dec_w0  = (const float*)d_in[7];
  const float* dec_b0  = (const float*)d_in[8];
  const float* dec_w1  = (const float*)d_in[9];
  const float* dec_b1  = (const float*)d_in[10];
  const float* dec_w2  = (const float*)d_in[11];
  const float* dec_b2  = (const float*)d_in[12];
  const float* attn_wi = (const float*)d_in[13];
  const float* attn_bi = (const float*)d_in[14];
  const float* attn_wh = (const float*)d_in[15];
  const float* attn_bh = (const float*)d_in[16];
  const float* attn_wv = (const float*)d_in[17];
  const float* attn_bv = (const float*)d_in[18];
  const float* tcnn_w  = (const float*)d_in[19];
  const float* tcnn_b  = (const float*)d_in[20];
  float* out = (float*)d_out;

  const int KT65 = 65 * 9, KC65 = (KT65 + 31) / 32;     // 585, 19
  const int KT128 = 128 * 9, KC128 = (KT128 + 31) / 32; // 1152, 36

  // workspace carve-out
  char* W = (char*)d_ws;
  size_t off = 0;
  auto alloc = [&](size_t bytes) -> void* {
    void* p = W + off;
    off = (off + bytes + 255) & ~(size_t)255;
    return p;
  };
  float* xs   = (float*)alloc((size_t)BB * TT * DD * SP * 4);
  float* z    = (float*)alloc((size_t)MG * COUT * 4);
  // guarded concat input for the WMMA GEMM (front/back guards absorb
  // clamp-free im2col taps that are masked out anyway)
  float* xing = (float*)alloc((size_t)(256 + BB * 128 * SP + SP + 256) * 4);
  float* xin  = xing + 256;
  // zero-initialized state block: h0,c0,zeros1,h1,c1,h2,c2
  float* h0   = (float*)alloc((size_t)BB * SP * 4);
  float* c0   = (float*)alloc((size_t)BB * SP * 4);
  float* zin  = (float*)alloc((size_t)BB * SP * 4);
  float* h1   = (float*)alloc((size_t)BB * HH * SP * 4);
  float* c1   = (float*)alloc((size_t)BB * HH * SP * 4);
  float* h2   = (float*)alloc((size_t)BB * HH * SP * 4);
  float* c2   = (float*)alloc((size_t)BB * HH * SP * 4);
  float* hid_e = (float*)alloc((size_t)BB * 32 * SP * 4);
  float* ebuf  = (float*)alloc((size_t)BB * 32 * SP * 4);
  float* alpha = (float*)alloc((size_t)BB * DD * SP * 4);
  unsigned* gmax = (unsigned*)alloc(256);
  float* ring  = (float*)alloc((size_t)BB * 11 * SP * 4);
  unsigned short* wb_e1 = (unsigned short*)alloc((size_t)KC65 * 8192 * 2);
  unsigned short* wb_e2 = (unsigned short*)alloc((size_t)KC128 * 8192 * 2);
  unsigned short* wb_d0 = (unsigned short*)alloc((size_t)KC65 * 8192 * 2);
  unsigned short* wb_d1 = (unsigned short*)alloc((size_t)KC128 * 8192 * 2);

  auto blocks = [](long n) { return (unsigned)((n + 255) / 256); };

  // 1) pack weights to bf16 B-fragment layout
  {
    int t65 = KC65 * 8192, t128 = KC128 * 8192;
    pack_wB<<<blocks(t65), 256, 0, stream>>>(enc_w1, 65, KT65, wb_e1, t65);
    pack_wB<<<blocks(t128), 256, 0, stream>>>(enc_w2, 128, KT128, wb_e2, t128);
    pack_wB<<<blocks(t65), 256, 0, stream>>>(dec_w0, 65, KT65, wb_d0, t65);
    pack_wB<<<blocks(t128), 256, 0, stream>>>(dec_w1, 128, KT128, wb_d1, t128);
  }

  // 2) xs <- x (mutated by attention), states <- 0, ring <- x[:, :, SEL]
  hipMemcpyAsync(xs, x, (size_t)BB * TT * DD * SP * 4, hipMemcpyDeviceToDevice, stream);
  {
    long nz = 3L * BB * SP + 4L * BB * HH * SP;  // contiguous h0..c2 block
    fill_f32<<<2048, 256, 0, stream>>>(h0, 0.f, nz);
    ring_init<<<blocks((long)BB * TT * SP), 256, 0, stream>>>(x, ring, BB * TT * SP);
  }

  const dim3 gemmGrid(MG / 128, COUT / 64);

  // helper: concat + WMMA GEMM + gates for one big ConvLSTM cell step
  auto bigCell = [&](const float* xt, int cinX, float* h, float* c,
                     const unsigned short* wb, int KC,
                     const float* bias) {
    int CIN = cinX + HH;
    concat_copy<<<blocks((long)BB * CIN * SP), 256, 0, stream>>>(
        xt, cinX, h, HH, xin, BB * CIN * SP);
    convlstm_gemm_bf16<<<gemmGrid, 256, 0, stream>>>(xin, CIN, wb, KC, z);
    lstm_gates<<<blocks((long)BB * HH * SP), 256, 0, stream>>>(
        z, bias, HH, 1, h, (long)HH * SP, c, BB * HH * SP);
  };

  // 3) encoder: interleave enc0 (with attention), enc1, enc2 per timestep
  for (int t = 0; t < TT; ++t) {
    // attention: hid_e = conv(concat(h0,c0), attn_wh) + attn_bh
    conv3x3_direct<<<blocks((long)BB * 32 * SP), 256, 0, stream>>>(
        h0, SP, SP, 1, c0, SP, SP, 1, attn_wh, attn_bh, 32, nullptr, 0,
        hid_e, 32L * SP, SP, BB * 32 * SP);
    for (int k = 0; k < DD; ++k) {
      // e = tanh(conv(xs[:,:,k], attn_wi) + attn_bi + hid_e)
      conv3x3_direct<<<blocks((long)BB * 32 * SP), 256, 0, stream>>>(
          xs + (long)k * SP, (long)TT * DD * SP, (long)DD * SP, TT,
          nullptr, 0, 0, 0, attn_wi, attn_bi, 32, hid_e, 1,
          ebuf, 32L * SP, SP, BB * 32 * SP);
      // alpha[:,k] = conv(e, attn_wv) + attn_bv
      conv3x3_direct<<<blocks((long)BB * SP), 256, 0, stream>>>(
          ebuf, 32L * SP, SP, 32, nullptr, 0, 0, 0, attn_wv, attn_bv, 1,
          nullptr, 0, alpha + (long)k * SP, (long)DD * SP, SP, BB * SP);
    }
    fill_u32<<<1, 32, 0, stream>>>(gmax, 0u, 1);
    exp_max_reduce<<<180, 256, 0, stream>>>(alpha, BB * DD * SP, gmax);
    attn_apply<<<blocks((long)BB * TT * DD * SP), 256, 0, stream>>>(
        xs, alpha, gmax, BB * TT * DD * SP);

    // enc0 cell (direct conv, Cin=6, Cout=4)
    conv3x3_direct<<<blocks((long)BB * 4 * SP), 256, 0, stream>>>(
        xs + (long)t * DD * SP, (long)TT * DD * SP, SP, DD,
        h0, SP, SP, 1, enc_w0, nullptr, 4, nullptr, 0,
        z, 4L * SP, SP, BB * 4 * SP);
    lstm_gates<<<blocks((long)BB * SP), 256, 0, stream>>>(
        z, enc_b0, 1, 0, h0, SP, c0, BB * SP);

    // enc1 cell (WMMA, Cin=1+64), enc2 cell (WMMA, Cin=64+64)
    bigCell(h0, 1, h1, c1, wb_e1, KC65, enc_b1);
    bigCell(h1, 64, h2, c2, wb_e2, KC128, enc_b2);
  }

  // 4) decoder: dec0 reuses (h2,c2), dec1 reuses (h1,c1), dec2 reuses (h0,c0)
  for (int t = 0; t < 5; ++t) {
    bigCell(zin, 1, h2, c2, wb_d0, KC65, dec_b0);   // dec0, input zeros
    bigCell(h2, 64, h1, c1, wb_d1, KC128, dec_b1);  // dec1
    // dec2 cell (direct conv, Cin=64+1, Cout=4); h_cell -> ring slot (t+10)%11
    conv3x3_direct<<<blocks((long)BB * 4 * SP), 256, 0, stream>>>(
        h1, (long)HH * SP, SP, HH, h0, SP, SP, 1, dec_w2, nullptr, 4,
        nullptr, 0, z, 4L * SP, SP, BB * 4 * SP);
    lstm_gates<<<blocks((long)BB * SP), 256, 0, stream>>>(
        z, dec_b2, 1, 0, ring + (long)((t + 10) % 11) * SP, 11L * SP, c0,
        BB * SP);
    // tcnn: h_out = relu(conv1x1(ring window)); carry h0 <- h_out; emit output
    tcnn_step<<<blocks((long)BB * SP), 256, 0, stream>>>(
        ring, tcnn_w, tcnn_b, t, h0, out + (long)t * SP, 5L * SP, BB * SP);
  }
}
